// SingleAttention_29343216566720
// MI455X (gfx1250) — compile-verified, matrix operand order checked by reading, not run on hardware
//
#include <hip/hip_runtime.h>
#include <hip/hip_bf16.h>

typedef __attribute__((ext_vector_type(16))) _Float16     v16h;
typedef __attribute__((ext_vector_type(8)))  float        v8f;
typedef __attribute__((ext_vector_type(4)))  unsigned int v4u;
typedef __attribute__((ext_vector_type(8)))  int          v8i;
typedef __attribute__((ext_vector_type(4)))  int          v4i;

#define ATTN_B 2
#define ATTN_H 16
#define ATTN_S 2048
#define ATTN_D 64
#define QTILE  16
#define NEG_INF_F (-1.0e9f)
#define SCALE_F   (0.125f)   // 1/sqrt(64)

#if __has_builtin(__builtin_amdgcn_tensor_load_to_lds) && __has_builtin(__builtin_amdgcn_s_wait_tensorcnt)
#define USE_TDM 1
#else
#define USE_TDM 0
#endif

#if USE_TDM
// Stage a 32x64 f32 V tile (8 KB, rows contiguous) into LDS via the Tensor
// Data Mover. D# built per CDNA5 ISA §8.3/§8.4. Call wave-uniformly.
__device__ __forceinline__ void stage_v_tile_tdm(const float* vtile, float* dst) {
    const unsigned long long ga = (unsigned long long)(uintptr_t)vtile;
    const unsigned lds = (unsigned)(uintptr_t)dst;
    v4u g0;
    g0[0] = 1u;                                   // count=1 user descriptor
    g0[1] = lds;                                  // lds_addr
    g0[2] = (unsigned)(ga & 0xFFFFFFFFu);         // global_addr[31:0]
    g0[3] = (unsigned)((ga >> 32) & 0x01FFFFFFu)  // global_addr[56:32]
          | 0x80000000u;                          // type=2 ("image")
    v8i g1;
    g1[0] = (2 << 16);   // data_size = 4 bytes; wg_mask=0 (not in cluster)
    g1[1] = (64 << 16);  // tensor_dim0 = 64 (lo16)
    g1[2] = (32 << 16);  // tensor_dim0 hi=0 | tensor_dim1 = 32
    g1[3] = (64 << 16);  // tensor_dim1 hi=0 | tile_dim0 = 64
    g1[4] = 32;          // tile_dim1 = 32; tile_dim2 = 0 (2D)
    g1[5] = 64;          // tensor_dim0_stride = 64 (lo32)
    g1[6] = 0;           // stride0 hi | stride1 lo (unused, 2D)
    g1[7] = 0;
    const v4i z4 = {0, 0, 0, 0};
#if defined(__clang_major__) && __clang_major__ >= 23
    const v8i z8 = {0, 0, 0, 0, 0, 0, 0, 0};
    __builtin_amdgcn_tensor_load_to_lds(g0, g1, z4, z4, z8, 0);
#else
    __builtin_amdgcn_tensor_load_to_lds(g0, g1, z4, z4, 0);
#endif
}
#endif

// Synchronous fallback staging: 256 threads x 2 float4 = 8 KB tile.
__device__ __forceinline__ void stage_v_tile_sync(const float* vtile, float* dst, int tid) {
    const float4* s = (const float4*)vtile;
    float4* d = (float4*)dst;
    d[tid]       = s[tid];
    d[tid + 256] = s[tid + 256];
}

// One workgroup (256 threads = 8 waves) handles one (b,h, 16-query-row) strip.
//   Phase 1: raw scaled scores S = scale * Q K^T      -> LDS sS[16][2048] f32
//   Phase 2: mask (coalesced) + row softmax in LDS, stream attn to HBM
//   Phase 3: out = P @ V with TDM/async double-buffered V tiles in LDS
__global__ __launch_bounds__(256)
void attention_fused_kernel(const float* __restrict__ Q,
                            const float* __restrict__ K,
                            const float* __restrict__ V,
                            const int*   __restrict__ mask,
                            float* __restrict__ out,
                            float* __restrict__ attn) {
    __shared__ float sS[QTILE * ATTN_S];        // 128 KB score/prob strip
    __shared__ float sV[2 * 32 * ATTN_D];       // 16 KB ping-pong V tiles
    __shared__ float red[QTILE * 16];           // reduction pad

    const int bh   = blockIdx.x / (ATTN_S / QTILE);   // fused b*H+h
    const int qt   = blockIdx.x % (ATTN_S / QTILE);
    const int q0   = qt * QTILE;
    const int tid  = threadIdx.x;
    const int lane = tid & 31;
    const int wave = tid >> 5;
    const int hrow = lane & 15;               // M (or N) index within tile
    const int kb   = (lane < 16) ? 0 : 8;     // K-subblock base per ISA A/B layout

    const size_t bhS = (size_t)bh * ATTN_S;

    // ---- Phase 1: scaled scores into LDS ------------------------------
    const float* Qrow = Q + (bhS + q0 + hrow) * ATTN_D;
    v16h a0, a1;
    #pragma unroll
    for (int j = 0; j < 8; ++j) {
        a0[j]     = (_Float16)Qrow[kb + j];
        a0[j + 8] = (_Float16)Qrow[16 + kb + j];
        a1[j]     = (_Float16)Qrow[32 + kb + j];
        a1[j + 8] = (_Float16)Qrow[48 + kb + j];
    }

    for (int t = 0; t < 16; ++t) {
        const int kt = wave * 16 + t;         // this wave's key tile
        const int n0 = kt * 16;
        const float* Krow = K + (bhS + n0 + hrow) * ATTN_D;
        if (t < 15) {   // speculative prefetch of next key tile
            __builtin_prefetch(K + (bhS + n0 + 16 + hrow) * ATTN_D, 0, 0);
        }
        v16h b0, b1;
        #pragma unroll
        for (int j = 0; j < 8; ++j) {
            b0[j]     = (_Float16)Krow[kb + j];
            b0[j + 8] = (_Float16)Krow[16 + kb + j];
            b1[j]     = (_Float16)Krow[32 + kb + j];
            b1[j + 8] = (_Float16)Krow[48 + kb + j];
        }
        v8f c = {};
        c = __builtin_amdgcn_wmma_f32_16x16x32_f16(false, a0, false, b0,
                                                   (short)0, c, false, false);
        c = __builtin_amdgcn_wmma_f32_16x16x32_f16(false, a1, false, b1,
                                                   (short)0, c, false, false);
        const int kcol = n0 + hrow;
        #pragma unroll
        for (int r = 0; r < 8; ++r) {
            const int m = r + ((lane < 16) ? 0 : 8);
            sS[m * ATTN_S + kcol] = c[r] * SCALE_F;
        }
    }
    __syncthreads();

    // ---- Phase 2: mask (coalesced) + row softmax + write attn ---------
    {
        const int r2 = tid >> 4;   // row 0..15
        const int s2 = tid & 15;   // 16 threads per row
        const int* mrow = mask + (size_t)(q0 + r2) * ATTN_S;

        float mx = -__builtin_inff();
        for (int i = s2; i < ATTN_S; i += 16) {
            float s = sS[r2 * ATTN_S + i];
            if (mrow[i] == 0) s = NEG_INF_F;     // coalesced mask read
            sS[r2 * ATTN_S + i] = s;
            mx = fmaxf(mx, s);
        }
        red[r2 * 16 + s2] = mx;
        __syncthreads();
        mx = -__builtin_inff();
        #pragma unroll
        for (int j = 0; j < 16; ++j) mx = fmaxf(mx, red[r2 * 16 + j]);
        __syncthreads();

        float sum = 0.0f;
        for (int i = s2; i < ATTN_S; i += 16) {
            const float e = __expf(sS[r2 * ATTN_S + i] - mx);
            sS[r2 * ATTN_S + i] = e;
            sum += e;
        }
        red[r2 * 16 + s2] = sum;
        __syncthreads();
        sum = 0.0f;
        #pragma unroll
        for (int j = 0; j < 16; ++j) sum += red[r2 * 16 + j];
        const float inv = 1.0f / sum;

        float* attnrow = attn + (bhS + q0 + r2) * ATTN_S;
        for (int i = s2; i < ATTN_S; i += 16) {
            const float p = sS[r2 * ATTN_S + i] * inv;
            sS[r2 * ATTN_S + i] = p;     // keep normalized probs for phase 3
            attnrow[i] = p;              // the one big HBM write
        }
    }
    __syncthreads();

    // ---- Phase 3: out = P @ V, double-buffered V tiles in LDS ---------
    // Pre-stage tile 0 into buffer 0.
#if USE_TDM
    if (wave == 0) {
        stage_v_tile_tdm(V + bhS * ATTN_D, sV);
        __builtin_amdgcn_s_wait_tensorcnt(0);
    }
#else
    stage_v_tile_sync(V + bhS * ATTN_D, sV, tid);
#endif
    __syncthreads();

    const int n0 = (wave & 3) * 16;      // 16-wide slice of D (waves 0..3)
    v8f c = {};
    for (int it = 0; it < 64; ++it) {
        const int cur = it & 1;
        // Stage next tile into the other buffer, overlapped with compute.
        if (it + 1 < 64) {
            const float* nxt = V + (bhS + (size_t)(it + 1) * 32) * ATTN_D;
#if USE_TDM
            if (wave == 0) stage_v_tile_tdm(nxt, sV + (cur ^ 1) * (32 * ATTN_D));
#else
            stage_v_tile_sync(nxt, sV + (cur ^ 1) * (32 * ATTN_D), tid);
#endif
        }
        if (wave < 4) {
            const int k0 = it * 32;
            const float* vb = sV + cur * (32 * ATTN_D);
            v16h a, b;
            #pragma unroll
            for (int j = 0; j < 8; ++j) {
                a[j]     = (_Float16)sS[hrow * ATTN_S + k0 + kb + j];
                a[j + 8] = (_Float16)sS[hrow * ATTN_S + k0 + 16 + kb + j];
                b[j]     = (_Float16)vb[(kb + j)      * ATTN_D + n0 + hrow];
                b[j + 8] = (_Float16)vb[(16 + kb + j) * ATTN_D + n0 + hrow];
            }
            c = __builtin_amdgcn_wmma_f32_16x16x32_f16(false, a, false, b,
                                                       (short)0, c, false, false);
        }
#if USE_TDM
        if (wave == 0) __builtin_amdgcn_s_wait_tensorcnt(0);
#endif
        __syncthreads();
    }
    if (wave < 4) {
        #pragma unroll
        for (int r = 0; r < 8; ++r) {
            const int m = r + ((lane < 16) ? 0 : 8);
            out[(bhS + q0 + m) * ATTN_D + n0 + hrow] = c[r];
        }
    }
}

extern "C" void kernel_launch(void* const* d_in, const int* in_sizes, int n_in,
                              void* d_out, int out_size, void* d_ws, size_t ws_size,
                              hipStream_t stream) {
    const float* Q    = (const float*)d_in[0];
    const float* Kp   = (const float*)d_in[1];
    const float* V    = (const float*)d_in[2];
    const int*   mask = (const int*)d_in[3];

    float* out  = (float*)d_out;
    float* attn = out + (size_t)ATTN_B * ATTN_H * ATTN_S * ATTN_D;

    const int nblocks = ATTN_B * ATTN_H * (ATTN_S / QTILE);   // 4096
    attention_fused_kernel<<<dim3(nblocks), dim3(256), 0, stream>>>(
        Q, Kp, V, mask, out, attn);
}
